// FlashCRA_31404800868928
// MI455X (gfx1250) — matrix operand order repaired and checked by reading, not compiled, not run on hardware
//
#include <hip/hip_runtime.h>
#include <hip/hip_bf16.h>

// ---------------------------------------------------------------------------
// FlashCRA for MI455X (gfx1250): bf16 WMMA GEMMs + fused-RoPE + flash attention
// + Tensor Data Mover (TDM) tile staging with hardware LDS row padding.
// ---------------------------------------------------------------------------

typedef __bf16 bf16_t;
typedef __attribute__((ext_vector_type(16))) __bf16        v16bf;
typedef __attribute__((ext_vector_type(8)))  float         v8f;
typedef __attribute__((ext_vector_type(4)))  unsigned int  u32x4;
typedef __attribute__((ext_vector_type(4)))  int           i32x4;
typedef __attribute__((ext_vector_type(8)))  int           i32x8;
typedef __attribute__((ext_vector_type(4)))  float         f32x4;

#define B_    2
#define L_    2048
#define D_    1024
#define H_    16
#define HD_   64
#define HALF_ 32

#define BM 64
#define BN 64
#define BK 32
#define ASTR 40   // padded LDS row stride (bf16 elems): 64B row + 16B pad = 80B
#define BSTR 40
#define TSTR 72   // attention LDS stride: 128B row + 16B pad = 144B

#if __has_builtin(__builtin_amdgcn_tensor_load_to_lds)
#define HAVE_TDM 1
#else
#define HAVE_TDM 0
#endif

static __device__ __forceinline__ v8f zero8() {
  v8f z = {0.f, 0.f, 0.f, 0.f, 0.f, 0.f, 0.f, 0.f};
  return z;
}

static __device__ __forceinline__ v16bf ld_frag16(const bf16_t* p0, const bf16_t* p1) {
  union { u32x4 u[2]; v16bf v; } c;
  c.u[0] = *reinterpret_cast<const u32x4*>(p0);
  c.u[1] = *reinterpret_cast<const u32x4*>(p1);
  return c.v;
}

// A fragment (16xK tile, row-major LDS). Lane l holds row m=l%16; for half
// h=l/16: elems 0..7 -> k = koff + 8h + e, elems 8..15 -> k = koff + 16 + 8h + e.
static __device__ __forceinline__ v16bf a_frag(const bf16_t* row, int koff, int hh) {
  const bf16_t* p = row + koff + hh * 8;
  return ld_frag16(p, p + 16);
}

// B fragment from K-contiguous (transposed) LDS tile: lane l holds col n=l%16;
// for h=l/16: elems 0..7 -> k = koff + 8h + e, elems 8..15 -> k = koff + 16 + 8h + e.
static __device__ __forceinline__ v16bf b_frag(const bf16_t* row, int koff, int hh) {
  const bf16_t* p = row + koff + hh * 8;
  return ld_frag16(p, p + 16);
}

static __device__ __forceinline__ v8f wmma_bf16(v16bf a, v16bf b, v8f c) {
  return __builtin_amdgcn_wmma_f32_16x16x32_bf16(
      /*neg_a=*/false, a, /*neg_b=*/false, b,
      /*c_mod=*/(short)0, c, /*reuse_a=*/false, /*reuse_b=*/false);
}

// ---------------------------------------------------------------------------
// Tensor Data Mover: DMA a 2D bf16 tile [tile_h rows x tile_w elems] from
// global (row stride = stride_e elems) into LDS at byte offset lds_off,
// with hardware row padding (pad_i: 3 => every 16 DWORDs, 4 => every 32
// DWORDs; pad_a: 3 => insert 4 DWORDs) so LDS rows land on our padded strides.
// D# layout per CDNA5 ISA.  Issued by one wave; EXEC is
// ignored by TDM; completion tracked with TENSORcnt.
// ---------------------------------------------------------------------------
static __device__ __forceinline__ void tdm_load_2d(unsigned lds_off, const void* gsrc,
                                                   int tile_w, int tile_h, long stride_e,
                                                   int pad_i, int pad_a) {
#if HAVE_TDM
  unsigned long long ga = (unsigned long long)gsrc;
  u32x4 g0 = { 1u,                                    // count=1, user mode
               lds_off,                               // lds_addr
               (unsigned)ga,                          // global_addr[31:0]
               (unsigned)((ga >> 32) & 0x01FFFFFFu) | (2u << 30) };  // addr[56:32]|type=2
  unsigned long long sb = (unsigned long long)stride_e;
  i32x8 g1 = { (int)((1u << 16) |                    // data_size = 2 bytes
                     (1u << 20) |                    // pad_enable
                     ((unsigned)pad_i << 22) | ((unsigned)pad_a << 25)),
               (int)(((unsigned)tile_w & 0xFFFFu) << 16),   // tensor_dim0[15:0]
               (int)(((unsigned)tile_h & 0xFFFFu) << 16),   // tensor_dim1[15:0]
               (int)(((unsigned)tile_w & 0xFFFFu) << 16),   // tile_dim0
               (int)((unsigned)tile_h & 0xFFFFu),           // tile_dim1 (tile_dim2=0)
               (int)(unsigned)sb,                           // dim0 stride [31:0]
               (int)((unsigned)(sb >> 32) & 0xFFFFu),       // dim0 stride [47:32]
               0 };
  i32x4 gz = { 0, 0, 0, 0 };
#if __clang_major__ >= 23
  i32x8 gz8 = { 0, 0, 0, 0, 0, 0, 0, 0 };
  __builtin_amdgcn_tensor_load_to_lds(g0, g1, gz, gz, gz8, 0);
#else
  __builtin_amdgcn_tensor_load_to_lds(g0, g1, gz, gz, 0);
#endif
#else
  (void)lds_off; (void)gsrc; (void)tile_w; (void)tile_h;
  (void)stride_e; (void)pad_i; (void)pad_a;
#endif
}

static __device__ __forceinline__ void tdm_wait() {
#if HAVE_TDM
  __builtin_amdgcn_s_wait_tensorcnt((short)0);
#endif
}

// ---------------------------------------------------------------------------
// Shared tiled-GEMM core: C(64x64) per block, 128 threads = 4 waves, each wave
// owns a 16(M) x 64(N) strip -> 4 accumulator tiles.  fp32 A is converted to
// bf16 while staging; bf16 A is staged by the TDM.  B (fp32 row-major) is
// staged transposed so B fragments read contiguous K.
// ---------------------------------------------------------------------------
template <int K, int N, bool ABF16>
static __device__ __forceinline__ void gemm_body(const void* Aptr,
                                                 const float* __restrict__ Bw,
                                                 bf16_t* __restrict__ As,
                                                 bf16_t* __restrict__ Bs,
                                                 v8f (&acc)[4]) {
  const float*  Af = (const float*)Aptr;
  const bf16_t* Ab = (const bf16_t*)Aptr;
  const int tid  = threadIdx.x;
  const int lane = tid & 31, wave = tid >> 5;
  const int hh = lane >> 4, nl = lane & 15;
  const int m0 = blockIdx.y * BM;
  const int n0 = blockIdx.x * BN;
  const unsigned As_off = (unsigned)(unsigned long long)(uintptr_t)As;

#pragma unroll
  for (int t = 0; t < 4; ++t) acc[t] = zero8();

  for (int k0 = 0; k0 < K; k0 += BK) {
    __syncthreads();
    // ---- stage A tile (64 x 32) -> bf16 LDS ----
    if constexpr (ABF16) {
#if HAVE_TDM
      if (wave == 0)   // TDM: 64 rows x 64B, +16B pad -> 80B LDS rows (ASTR)
        tdm_load_2d(As_off, Ab + (long)m0 * K + k0, BK, BM, K, /*pad_i=*/3, /*pad_a=*/3);
#else
      u32x4 ar[2];
#pragma unroll
      for (int i = 0; i < 2; ++i) {
        int q8 = i * 128 + tid;
        ar[i] = *reinterpret_cast<const u32x4*>(Ab + ((long)m0 + (q8 >> 2)) * K + k0 + (q8 & 3) * 8);
      }
#pragma unroll
      for (int i = 0; i < 2; ++i) {
        int q8 = i * 128 + tid;
        *reinterpret_cast<u32x4*>(As + (q8 >> 2) * ASTR + (q8 & 3) * 8) = ar[i];
      }
#endif
    } else {
      // loads first (4 b128 loads in flight), then convert+store
      f32x4 ar[4];
#pragma unroll
      for (int i = 0; i < 4; ++i) {
        int q4 = i * 128 + tid;                 // 512 quads of 4 floats
        const float* g = (const float*)__builtin_assume_aligned(
            Af + ((long)m0 + (q4 >> 3)) * K + k0 + (q4 & 7) * 4, 16);
        ar[i] = *reinterpret_cast<const f32x4*>(g);
      }
#pragma unroll
      for (int i = 0; i < 4; ++i) {
        int q4 = i * 128 + tid;
        bf16_t* d = As + (q4 >> 3) * ASTR + (q4 & 7) * 4;
        d[0] = (bf16_t)ar[i].x; d[1] = (bf16_t)ar[i].y;
        d[2] = (bf16_t)ar[i].z; d[3] = (bf16_t)ar[i].w;
      }
    }
    // ---- stage B tile (32 x 64), transposed -> Bs[n][k] ----
    f32x4 br[4];
#pragma unroll
    for (int i = 0; i < 4; ++i) {
      int q4 = i * 128 + tid;                   // 512 quads of 4 floats
      const float* g = (const float*)__builtin_assume_aligned(
          Bw + (long)(k0 + (q4 >> 4)) * N + n0 + (q4 & 15) * 4, 16);
      br[i] = *reinterpret_cast<const f32x4*>(g);
    }
#pragma unroll
    for (int i = 0; i < 4; ++i) {
      int q4 = i * 128 + tid;
      int kk = q4 >> 4, n = (q4 & 15) * 4;
      Bs[(n + 0) * BSTR + kk] = (bf16_t)br[i].x;
      Bs[(n + 1) * BSTR + kk] = (bf16_t)br[i].y;
      Bs[(n + 2) * BSTR + kk] = (bf16_t)br[i].z;
      Bs[(n + 3) * BSTR + kk] = (bf16_t)br[i].w;
    }
    if (k0 + BK < K) {   // pull next tiles toward the caches (global_prefetch_b8)
      if constexpr (ABF16)
        __builtin_prefetch(&Ab[((long)m0 + (tid >> 1)) * K + k0 + BK], 0, 1);
      else
        __builtin_prefetch(&Af[((long)m0 + (tid >> 1)) * K + k0 + BK], 0, 1);
      __builtin_prefetch(&Bw[(long)(k0 + BK + (tid >> 2)) * N + n0], 0, 1);
    }
    if constexpr (ABF16) { if (wave == 0) tdm_wait(); }
    __syncthreads();
    // ---- compute: preload all B frags, then back-to-back WMMAs ----
    v16bf a = a_frag(As + (16 * wave + nl) * ASTR, 0, hh);
    v16bf bf4[4];
#pragma unroll
    for (int t = 0; t < 4; ++t) bf4[t] = b_frag(Bs + (t * 16 + nl) * BSTR, 0, hh);
#pragma unroll
    for (int t = 0; t < 4; ++t) acc[t] = wmma_bf16(a, bf4[t], acc[t]);
  }
}

// ---------------------------------------------------------------------------
// Kernel 1: phases = p @ proj   (fp32 out, [B*L, H*HALF])
// ---------------------------------------------------------------------------
__global__ __launch_bounds__(128) void k_phase_gemm(const float* __restrict__ p,
                                                    const float* __restrict__ proj,
                                                    float* __restrict__ phases) {
  __shared__ __align__(16) bf16_t As[BM * ASTR];
  __shared__ __align__(16) bf16_t Bs[BN * BSTR];
  v8f acc[4];
  gemm_body<D_, D_ / 2, false>(p, proj, As, Bs, acc);
  const int lane = threadIdx.x & 31, wave = threadIdx.x >> 5;
  const int hh = lane >> 4, nl = lane & 15;
  const int m0 = blockIdx.y * BM, n0 = blockIdx.x * BN;
#pragma unroll
  for (int t = 0; t < 4; ++t) {
#pragma unroll
    for (int r = 0; r < 8; ++r) {
      int m = m0 + 16 * wave + r + 8 * hh;
      phases[m * (D_ / 2) + n0 + t * 16 + nl] = acc[t][r];
    }
  }
}

// ---------------------------------------------------------------------------
// Kernel 2: qkv = x @ Wqkv + b, RoPE fused into the epilogue.  BN=64 == HD, so
// each block's N-tile is exactly one (q/k/v, head) slab and rotate-half pairs
// (hd, hd+32) live in C-tiles (0,2)/(1,3) at the same lane/element.
// Writes bf16 q/k/v in [B,H,L,HD].
// ---------------------------------------------------------------------------
__global__ __launch_bounds__(128) void k_qkv_rope(const float* __restrict__ x,
                                                  const float* __restrict__ Wqkv,
                                                  const float* __restrict__ bqkv,
                                                  const float* __restrict__ phases,
                                                  bf16_t* __restrict__ qws,
                                                  bf16_t* __restrict__ kws,
                                                  bf16_t* __restrict__ vws) {
  __shared__ __align__(16) bf16_t As[BM * ASTR];
  __shared__ __align__(16) bf16_t Bs[BN * BSTR];
  v8f acc[4];
  gemm_body<D_, 3 * D_, false>(x, Wqkv, As, Bs, acc);
  const int lane = threadIdx.x & 31, wave = threadIdx.x >> 5;
  const int hh = lane >> 4, nl = lane & 15;
  const int m0 = blockIdx.y * BM, n0 = blockIdx.x * BN;
  const int which = n0 / D_;                 // 0=q 1=k 2=v
  const int head = (n0 % D_) / HD_;
  float bias[4];
#pragma unroll
  for (int t = 0; t < 4; ++t) bias[t] = bqkv[n0 + t * 16 + nl];

  if (which == 2) {                          // V: bias + cast, no rotation
#pragma unroll
    for (int t = 0; t < 4; ++t) {
#pragma unroll
      for (int r = 0; r < 8; ++r) {
        int mg = m0 + 16 * wave + r + 8 * hh;
        int bb = mg >> 11, tok = mg & (L_ - 1);
        long base = ((long)(bb * H_ + head) * L_ + tok) << 6;
        vws[base + t * 16 + nl] = (bf16_t)(acc[t][r] + bias[t]);
      }
    }
  } else {                                   // Q/K: rotate-half with phases
    bf16_t* dst = which ? kws : qws;
#pragma unroll
    for (int pt = 0; pt < 2; ++pt) {         // pairs (0,2) and (1,3)
      int j = pt * 16 + nl;                  // 0..31
#pragma unroll
      for (int r = 0; r < 8; ++r) {
        int mg = m0 + 16 * wave + r + 8 * hh;
        int bb = mg >> 11, tok = mg & (L_ - 1);
        float ph = phases[mg * (D_ / 2) + head * HALF_ + j];
        float sn, cs;
        __sincosf(ph, &sn, &cs);
        float t1 = acc[pt][r]     + bias[pt];
        float t2 = acc[pt + 2][r] + bias[pt + 2];
        long base = ((long)(bb * H_ + head) * L_ + tok) << 6;
        dst[base + j]         = (bf16_t)(t1 * cs - t2 * sn);
        dst[base + HALF_ + j] = (bf16_t)(t1 * sn + t2 * cs);
      }
    }
  }
}

// ---------------------------------------------------------------------------
// Kernel 3: flash attention.  grid = (L/64 query tiles, B*H heads).
// 4 waves x 16 query rows; 64-key blocks streamed through LDS.  Q and K tiles
// are DMA'd by the TDM (64x128B rows, hardware-padded to 144B); V is staged
// transposed by the waves; P transposes through per-wave LDS.
// ---------------------------------------------------------------------------
__global__ __launch_bounds__(128) void k_attn(const bf16_t* __restrict__ q,
                                              const bf16_t* __restrict__ k,
                                              const bf16_t* __restrict__ v,
                                              bf16_t* __restrict__ o) {
  __shared__ __align__(16) bf16_t Qs[64 * TSTR];
  __shared__ __align__(16) bf16_t Ks[64 * TSTR];       // [key][hd]
  __shared__ __align__(16) bf16_t Vt[64 * TSTR];       // [hd][key] (transposed)
  __shared__ __align__(16) bf16_t Ps[4 * 16 * TSTR];   // per-wave P transpose buf
  const int tid = threadIdx.x, lane = tid & 31, wave = tid >> 5;
  const int hh = lane >> 4, nl = lane & 15;
  const int bh = blockIdx.y;
  const int l0 = blockIdx.x * 64;
  const long qb = ((long)bh * L_ + l0) << 6;
  const unsigned Qs_off = (unsigned)(unsigned long long)(uintptr_t)Qs;
  const unsigned Ks_off = (unsigned)(unsigned long long)(uintptr_t)Ks;

  // stage Q tile (64 x 64 bf16) once via TDM
#if HAVE_TDM
  if (wave == 0) tdm_load_2d(Qs_off, q + qb, HD_, 64, HD_, /*pad_i=*/4, /*pad_a=*/3);
#else
#pragma unroll
  for (int i = 0; i < 4; ++i) {
    int q8 = i * 128 + tid;
    const u32x4* g = reinterpret_cast<const u32x4*>(q + qb + q8 * 8);
    *reinterpret_cast<u32x4*>(Qs + (q8 >> 3) * TSTR + (q8 & 7) * 8) = *g;
  }
#endif

  v8f O[4];
#pragma unroll
  for (int t = 0; t < 4; ++t) O[t] = zero8();
  float mrow[8], lrow[8];
#pragma unroll
  for (int r = 0; r < 8; ++r) { mrow[r] = -1e30f; lrow[r] = 0.f; }

  bf16_t* Pw = Ps + wave * 16 * TSTR;
  const bf16_t* qrow = Qs + (16 * wave + nl) * TSTR;
  const float SC = 0.125f;                   // HD^-0.5

  for (int kb = 0; kb < L_ / 64; ++kb) {
    __syncthreads();
    const long kvb = ((long)bh * L_ + kb * 64) << 6;
    // K tile by TDM; V tile staged transposed by the waves (regs first)
#if HAVE_TDM
    if (wave == 0) tdm_load_2d(Ks_off, k + kvb, HD_, 64, HD_, /*pad_i=*/4, /*pad_a=*/3);
#else
#pragma unroll
    for (int i = 0; i < 4; ++i) {
      int q8 = i * 128 + tid;
      const u32x4* g = reinterpret_cast<const u32x4*>(k + kvb + q8 * 8);
      *reinterpret_cast<u32x4*>(Ks + (q8 >> 3) * TSTR + (q8 & 7) * 8) = *g;
    }
#endif
    u32x4 vr[4];
#pragma unroll
    for (int i = 0; i < 4; ++i) {
      int q8 = i * 128 + tid;
      vr[i] = *reinterpret_cast<const u32x4*>(v + kvb + q8 * 8);
    }
#pragma unroll
    for (int i = 0; i < 4; ++i) {
      int q8 = i * 128 + tid;
      int row = q8 >> 3, c0 = (q8 & 7) * 8;
      union { u32x4 u; bf16_t h[8]; } vv;
      vv.u = vr[i];
#pragma unroll
      for (int j = 0; j < 8; ++j) Vt[(c0 + j) * TSTR + row] = vv.h[j];
    }
    if (kb + 1 < L_ / 64)
      __builtin_prefetch(k + kvb + 4096 + tid * 32, 0, 1);
#if HAVE_TDM
    if (wave == 0) tdm_wait();
#endif
    __syncthreads();

    // ---- S = Q K^T  (HD=64 -> 2 K-chunks of 32) ----
    v8f S[4];
#pragma unroll
    for (int t = 0; t < 4; ++t) S[t] = zero8();
#pragma unroll
    for (int c = 0; c < 2; ++c) {
      v16bf a = a_frag(qrow, c * 32, hh);
      v16bf bf4[4];
#pragma unroll
      for (int t = 0; t < 4; ++t) bf4[t] = b_frag(Ks + (t * 16 + nl) * TSTR, c * 32, hh);
#pragma unroll
      for (int t = 0; t < 4; ++t) S[t] = wmma_bf16(a, bf4[t], S[t]);
    }

    // ---- online softmax (rows replicated across the 16-lane half) ----
#pragma unroll
    for (int r = 0; r < 8; ++r) {
      float s0 = S[0][r] * SC, s1 = S[1][r] * SC;
      float s2 = S[2][r] * SC, s3 = S[3][r] * SC;
      float mx = fmaxf(fmaxf(s0, s1), fmaxf(s2, s3));
#pragma unroll
      for (int off = 1; off < 16; off <<= 1) mx = fmaxf(mx, __shfl_xor(mx, off, 32));
      float mnew  = fmaxf(mrow[r], mx);
      float alpha = __expf(mrow[r] - mnew);
      mrow[r] = mnew;
      float p0 = __expf(s0 - mnew), p1 = __expf(s1 - mnew);
      float p2 = __expf(s2 - mnew), p3 = __expf(s3 - mnew);
      float sum = p0 + p1 + p2 + p3;
#pragma unroll
      for (int off = 1; off < 16; off <<= 1) sum += __shfl_xor(sum, off, 32);
      lrow[r] = lrow[r] * alpha + sum;
#pragma unroll
      for (int t = 0; t < 4; ++t) O[t][r] *= alpha;
      int prow = (r + 8 * hh) * TSTR;        // transpose P into A-layout
      Pw[prow + 0 * 16 + nl] = (bf16_t)p0;
      Pw[prow + 1 * 16 + nl] = (bf16_t)p1;
      Pw[prow + 2 * 16 + nl] = (bf16_t)p2;
      Pw[prow + 3 * 16 + nl] = (bf16_t)p3;
    }

    // ---- O += P V  (64 keys -> 2 K-chunks of 32) ----
#pragma unroll
    for (int c = 0; c < 2; ++c) {
      v16bf a = a_frag(Pw + nl * TSTR, c * 32, hh);
      v16bf bf4[4];
#pragma unroll
      for (int t = 0; t < 4; ++t) bf4[t] = b_frag(Vt + (t * 16 + nl) * TSTR, c * 32, hh);
#pragma unroll
      for (int t = 0; t < 4; ++t) O[t] = wmma_bf16(a, bf4[t], O[t]);
    }
  }

  // ---- normalize and emit bf16 attn-out in [B, L, D] ----
  const int bb = bh / H_, head = bh % H_;
#pragma unroll
  for (int t = 0; t < 4; ++t) {
#pragma unroll
    for (int r = 0; r < 8; ++r) {
      int tok = l0 + 16 * wave + r + 8 * hh;
      float ov = O[t][r] / lrow[r];
      o[((long)(bb * L_ + tok) * D_) + head * HD_ + t * 16 + nl] = (bf16_t)ov;
    }
  }
}

// ---------------------------------------------------------------------------
// Kernel 4: out = attn @ Wout + bout   (fp32 out); A tiles staged by TDM
// ---------------------------------------------------------------------------
__global__ __launch_bounds__(128) void k_out_gemm(const bf16_t* __restrict__ a,
                                                  const float* __restrict__ Wout,
                                                  const float* __restrict__ bout,
                                                  float* __restrict__ out) {
  __shared__ __align__(16) bf16_t As[BM * ASTR];
  __shared__ __align__(16) bf16_t Bs[BN * BSTR];
  v8f acc[4];
  gemm_body<D_, D_, true>(a, Wout, As, Bs, acc);
  const int lane = threadIdx.x & 31, wave = threadIdx.x >> 5;
  const int hh = lane >> 4, nl = lane & 15;
  const int m0 = blockIdx.y * BM, n0 = blockIdx.x * BN;
#pragma unroll
  for (int t = 0; t < 4; ++t) {
    float bv = bout[n0 + t * 16 + nl];
#pragma unroll
    for (int r = 0; r < 8; ++r) {
      int m = m0 + 16 * wave + r + 8 * hh;
      out[(long)m * D_ + n0 + t * 16 + nl] = acc[t][r] + bv;
    }
  }
}

// ---------------------------------------------------------------------------
extern "C" void kernel_launch(void* const* d_in, const int* in_sizes, int n_in,
                              void* d_out, int out_size, void* d_ws, size_t ws_size,
                              hipStream_t stream) {
  (void)in_sizes; (void)n_in; (void)out_size; (void)ws_size;
  const float* x    = (const float*)d_in[0];
  const float* p    = (const float*)d_in[1];
  const float* Wqkv = (const float*)d_in[2];
  const float* bqkv = (const float*)d_in[3];
  const float* Wout = (const float*)d_in[4];
  const float* bout = (const float*)d_in[5];
  const float* proj = (const float*)d_in[6];
  float* out = (float*)d_out;

  // workspace: phases fp32 (8MB) | q,k,v bf16 [B,H,L,HD] (8MB each) | attn bf16 (8MB)
  char* ws = (char*)d_ws;
  float*  phases = (float*)ws;
  bf16_t* qws = (bf16_t*)(ws + (size_t)8  * 1024 * 1024);
  bf16_t* kws = (bf16_t*)(ws + (size_t)16 * 1024 * 1024);
  bf16_t* vws = (bf16_t*)(ws + (size_t)24 * 1024 * 1024);
  bf16_t* aws = (bf16_t*)(ws + (size_t)32 * 1024 * 1024);

  k_phase_gemm<<<dim3((D_ / 2) / BN, (B_ * L_) / BM), 128, 0, stream>>>(p, proj, phases);
  k_qkv_rope <<<dim3((3 * D_) / BN, (B_ * L_) / BM), 128, 0, stream>>>(x, Wqkv, bqkv, phases,
                                                                       qws, kws, vws);
  k_attn     <<<dim3(L_ / 64, B_ * H_), 128, 0, stream>>>(qws, kws, vws, aws);
  k_out_gemm <<<dim3(D_ / BN, (B_ * L_) / BM), 128, 0, stream>>>(aws, Wout, bout, out);
}
